// Mug_head_49143015801416
// MI455X (gfx1250) — matrix operand order repaired and checked by reading, not compiled, not run on hardware
//
#include <hip/hip_runtime.h>
#include <hip/hip_bf16.h>
#include <math.h>

typedef __attribute__((ext_vector_type(16))) _Float16 v16h;
typedef __attribute__((ext_vector_type(8)))  _Float16 h8;
typedef __attribute__((ext_vector_type(4)))  _Float16 h4;
typedef __attribute__((ext_vector_type(8)))  float    v8f;

#define A_N 256
#define L_N 32
#define B_N 256
#define V_N 16
#define D_N 512
#define TAU_F 100.0f

// ---------------------------------------------------------------------------
// Kernel 1: L2-normalize rows of a [nrows, 512] f32 matrix into f16.
// One wave (32 lanes) per row; each lane handles 4 float4 chunks (16 elems).
// ---------------------------------------------------------------------------
__global__ __launch_bounds__(256) void norm_rows_f32_to_f16(
    const float* __restrict__ in, _Float16* __restrict__ out, int nrows)
{
    int row  = blockIdx.x * (blockDim.x >> 5) + (threadIdx.x >> 5);
    if (row >= nrows) return;            // uniform per-wave
    int lane = threadIdx.x & 31;

    const float4* p = (const float4*)(in + (size_t)row * D_N);
    float4 x0 = p[lane];
    float4 x1 = p[lane + 32];
    float4 x2 = p[lane + 64];
    float4 x3 = p[lane + 96];

    float ss = x0.x*x0.x + x0.y*x0.y + x0.z*x0.z + x0.w*x0.w
             + x1.x*x1.x + x1.y*x1.y + x1.z*x1.z + x1.w*x1.w
             + x2.x*x2.x + x2.y*x2.y + x2.z*x2.z + x2.w*x2.w
             + x3.x*x3.x + x3.y*x3.y + x3.z*x3.z + x3.w*x3.w;
    #pragma unroll
    for (int off = 16; off; off >>= 1) ss += __shfl_xor(ss, off, 32);

    float inv = 1.0f / fmaxf(sqrtf(ss), 1e-6f);

    _Float16* o = out + (size_t)row * D_N;
    h4 y;
    y[0] = (_Float16)(x0.x*inv); y[1] = (_Float16)(x0.y*inv);
    y[2] = (_Float16)(x0.z*inv); y[3] = (_Float16)(x0.w*inv);
    *(h4*)(o + lane*4)       = y;
    y[0] = (_Float16)(x1.x*inv); y[1] = (_Float16)(x1.y*inv);
    y[2] = (_Float16)(x1.z*inv); y[3] = (_Float16)(x1.w*inv);
    *(h4*)(o + 128 + lane*4) = y;
    y[0] = (_Float16)(x2.x*inv); y[1] = (_Float16)(x2.y*inv);
    y[2] = (_Float16)(x2.z*inv); y[3] = (_Float16)(x2.w*inv);
    *(h4*)(o + 256 + lane*4) = y;
    y[0] = (_Float16)(x3.x*inv); y[1] = (_Float16)(x3.y*inv);
    y[2] = (_Float16)(x3.z*inv); y[3] = (_Float16)(x3.w*inv);
    *(h4*)(o + 384 + lane*4) = y;
}

// ---------------------------------------------------------------------------
// Kernel 2: fused WMMA GEMM + two-level softmax pooling.
// Block = 256 threads (8 wave32). Block handles 1 text `a` x 4 videos `b`.
// Wave w: row_half = w&1 (tokens 0-15 or 16-31), b_local = w>>1.
// Each wave computes a 16x16 r-tile via v_wmma_f32_16x16x32_f16, K=512.
// r blocks staged in LDS; waves 0-3 then pool one (a,b) pair each.
// ---------------------------------------------------------------------------
__global__ __launch_bounds__(256) void mug_head_kernel(
    const _Float16* __restrict__ tfh,   // [A*L, 512] normalized f16
    const _Float16* __restrict__ vfh,   // [B*V, 512] normalized f16
    const float*    __restrict__ mask,  // [A, L]
    float*          __restrict__ out)   // [A, B]
{
    __shared__ float rbuf[4][L_N][V_N + 1];   // +1 pad: 8704 B

    const int a      = blockIdx.y;
    const int b_base = blockIdx.x * 4;
    const int wave   = threadIdx.x >> 5;
    const int lane   = threadIdx.x & 31;
    const int half   = wave & 1;              // which 16-token half
    const int bl     = wave >> 1;             // which of the 4 videos
    const int b      = b_base + bl;

    const int mrow = lane & 15;               // A-row / B-col index within tile
    const int hiK  = lane >> 4;               // lane group selector

    // A-layout (16-bit 16x32): lanes 0-15 hold K {0..15},
    // lanes 16-31 hold K {16..31} of row M = lane&15 (matches B).
    const _Float16* arow = tfh + (size_t)(a * L_N + half * 16 + mrow) * D_N;
    const int aoff = hiK ? 16 : 0;
    // B-layout used here (32x16): lane n holds column N=n, K contiguous;
    // lanes 16-31 at K+16.  vfh rows are columns of B -> contiguous loads.
    const _Float16* bcol = vfh + (size_t)(b * V_N + mrow) * D_N;
    const int boff = hiK ? 16 : 0;

    v8f c = {};
    #pragma unroll 4
    for (int k0 = 0; k0 < D_N; k0 += 32) {
        h8 alo = *(const h8*)(arow + k0 + aoff);
        h8 ahi = *(const h8*)(arow + k0 + aoff + 8);
        v16h af = __builtin_shufflevector(alo, ahi,
                    0,1,2,3,4,5,6,7,8,9,10,11,12,13,14,15);
        v16h bf = *(const v16h*)(bcol + k0 + boff);
        c = __builtin_amdgcn_wmma_f32_16x16x32_f16(
                false, af, false, bf, (short)0, c, false, false);
    }

    // C layout: lanes 0-15 hold M=0..7 (VGPR i -> M=i), lanes 16-31 M=8..15;
    // column N = lane&15.  Apply text mask, stage to LDS.
    #pragma unroll
    for (int i = 0; i < 8; ++i) {
        int t = half * 16 + hiK * 8 + i;
        rbuf[bl][t][mrow] = c[i] * mask[a * L_N + t];
    }
    __syncthreads();

    if (wave < 4) {
        const int bb = b_base + wave;

        // ---- frame-axis softmax pooling: lane t = 0..31 handles one row
        float m1 = -INFINITY;
        #pragma unroll
        for (int v = 0; v < V_N; ++v) m1 = fmaxf(m1, rbuf[wave][lane][v]);
        float s1 = 0.f, n1 = 0.f;
        #pragma unroll
        for (int v = 0; v < V_N; ++v) {
            float r = rbuf[wave][lane][v];
            float e = __expf(TAU_F * (r - m1));
            s1 += e; n1 += r * e;
        }
        float t2v = n1 / s1;                       // t2v[a,bb,t=lane]

        // ---- token-axis masked softmax pooling: lane handles column v=lane&15
        int v = lane & 15;
        float m2 = -INFINITY;
        for (int t = 0; t < L_N; ++t) {
            float r  = rbuf[wave][t][v];
            float rp = (r == 0.0f) ? -INFINITY : r;
            m2 = fmaxf(m2, rp);
        }
        float s2 = 0.f, n2 = 0.f;
        for (int t = 0; t < L_N; ++t) {
            float r  = rbuf[wave][t][v];
            float rp = (r == 0.0f) ? -INFINITY : r;
            float e  = __expf(TAU_F * (rp - m2));
            s2 += e; n2 += r * e;
        }
        float v2t = n2 / s2;                       // v2t[a,bb,v] (dup on hi lanes)

        // ---- second round: masked softmax over 32 tokens (full wave)
        float tp = (t2v == 0.0f) ? -INFINITY : t2v;
        float m3 = tp;
        #pragma unroll
        for (int off = 16; off; off >>= 1) m3 = fmaxf(m3, __shfl_xor(m3, off, 32));
        float e3 = __expf(TAU_F * (tp - m3));
        float s3 = e3, n3 = t2v * e3;
        #pragma unroll
        for (int off = 16; off; off >>= 1) {
            s3 += __shfl_xor(s3, off, 32);
            n3 += __shfl_xor(n3, off, 32);
        }
        float t2v_ab = n3 / s3;

        // ---- second round: plain softmax over 16 frames (group of 16 lanes)
        float m4 = v2t;
        #pragma unroll
        for (int off = 8; off; off >>= 1) m4 = fmaxf(m4, __shfl_xor(m4, off, 32));
        float e4 = __expf(TAU_F * (v2t - m4));
        float s4 = e4, n4 = v2t * e4;
        #pragma unroll
        for (int off = 8; off; off >>= 1) {
            s4 += __shfl_xor(s4, off, 32);
            n4 += __shfl_xor(n4, off, 32);
        }
        float v2t_ab = n4 / s4;

        if (lane == 0)
            out[(size_t)a * B_N + bb] = 0.5f * (t2v_ab + v2t_ab);
    }
}

extern "C" void kernel_launch(void* const* d_in, const int* in_sizes, int n_in,
                              void* d_out, int out_size, void* d_ws, size_t ws_size,
                              hipStream_t stream) {
    const float* text  = (const float*)d_in[0];   // [256,32,512] f32
    const float* video = (const float*)d_in[1];   // [256,16,512] f32
    const float* mask  = (const float*)d_in[2];   // [256,32]     f32
    float*       out   = (float*)d_out;           // [256,256]    f32

    _Float16* tfh = (_Float16*)d_ws;                          // 8 MB
    _Float16* vfh = tfh + (size_t)A_N * L_N * D_N;            // +4 MB

    const int trows = A_N * L_N;   // 8192
    const int vrows = B_N * V_N;   // 4096
    norm_rows_f32_to_f16<<<(trows + 7) / 8, 256, 0, stream>>>(text,  tfh, trows);
    norm_rows_f32_to_f16<<<(vrows + 7) / 8, 256, 0, stream>>>(video, vfh, vrows);

    dim3 grid(B_N / 4, A_N);       // (64, 256) blocks
    mug_head_kernel<<<grid, 256, 0, stream>>>(tfh, vfh, mask, out);
}
